// CustomAttention_38809324487461
// MI455X (gfx1250) — matrix operand order repaired and checked by reading, not compiled, not run on hardware
//
#include <hip/hip_runtime.h>
#include <hip/hip_bf16.h>
#include <math.h>

typedef __attribute__((ext_vector_type(16))) __bf16 v16bf;
typedef __attribute__((ext_vector_type(8)))  float  v8f;

#define D_MODEL 1024
#define N_HEADS 16
#define D_HEAD  64
#define SEQ     4096
#define BATCH   4
#define MROWS   (BATCH * SEQ)      // 16384
#define NCOLS   (3 * D_MODEL)      // 3072

// ---------------------------------------------------------------------------
// CDNA5 async global->LDS copy (ASYNCcnt-tracked, no VGPR round trip).
// GVS mode: vdst = per-lane LDS byte offset, vaddr = per-lane 32-bit global
// byte offset, saddr = uniform 64-bit base.  (ISA ch10 async instructions)
__device__ __forceinline__ void async_ld_b128(void* lds_dst, const void* gbase,
                                              unsigned goff_bytes) {
  unsigned l = (unsigned)(size_t)lds_dst;   // low 32 bits of flat = LDS offset
  asm volatile("global_load_async_to_lds_b128 %0, %1, %2"
               :
               : "v"(l), "v"(goff_bytes), "s"(gbase)
               : "memory");
}
__device__ __forceinline__ void wait_async0() {
  asm volatile("s_wait_asynccnt 0x0" ::: "memory");
}

// ---------------------------------------------------------------------------
// WMMA operand loaders (wave32); A and B both use contiguous k order, k = 16*(lane>>4) + s.
// A operand, 16x32 bf16, source tile laid out [M][K] with K contiguous.
__device__ __forceinline__ v16bf load_a16x32(const __bf16* t, int r0, int ld,
                                             int k0, int lane) {
  int m  = lane & 15;
  int kh = (lane >> 4) << 4;
  union { v16bf v; unsigned u[8]; } r;
  const __bf16* p = t + (size_t)(r0 + m) * ld + k0;
#pragma unroll
  for (int i = 0; i < 8; ++i) {
    int kb = kh + (i << 1);   // 16*(lane>>4) + 2*i
    r.u[i] = *reinterpret_cast<const unsigned*>(p + kb);
  }
  return r.v;
}

// B operand, 32x16 bf16, B[k][n] sourced from a tile laid out [N][K], K contig.
__device__ __forceinline__ v16bf load_b32x16(const __bf16* t, int n0, int ld,
                                             int k0, int lane) {
  int n  = lane & 15;
  int kb = (lane >> 4) << 4;
  union { v16bf v; unsigned u[8]; } r;
  const __bf16* p = t + (size_t)(n0 + n) * ld + k0 + kb;
#pragma unroll
  for (int i = 0; i < 8; ++i)
    r.u[i] = *reinterpret_cast<const unsigned*>(p + (i << 1));
  return r.v;
}

// B operand for probs@v straight from global qkv storage: v[e][s] at
// vb + s*3072 + e, e contiguous -> each lane reads 32 contiguous bytes.
__device__ __forceinline__ v16bf load_bv(const __bf16* vb, int s0, int half,
                                         int lane) {
  int s  = s0 + (lane & 15);
  int eb = half * 32 + ((lane >> 4) << 4);
  union { v16bf v; uint4 q[2]; } r;
  const __bf16* p = vb + (size_t)s * NCOLS + eb;
  r.q[0] = *reinterpret_cast<const uint4*>(p);
  r.q[1] = *reinterpret_cast<const uint4*>(p + 8);
  return r.v;
}

// ---------------------------------------------------------------------------
__global__ void cast_f32_bf16(const float* __restrict__ in,
                              __bf16* __restrict__ out, int n) {
  int i = blockIdx.x * blockDim.x + threadIdx.x;
  if (i < n) out[i] = (__bf16)in[i];
}

// ---------------------------------------------------------------------------
// QKV GEMM: C[m][n] = sum_k A[m][k] * W[n][k] + bias[n].   M=16384 N=3072 K=1024
// Block: 256 threads (8 waves). Block tile 256(M) x 64(N), K-step 32.
// Double-buffered async global->LDS staging overlapped with WMMA compute.
__global__ __launch_bounds__(256)
void qkv_gemm(const __bf16* __restrict__ A, const __bf16* __restrict__ W,
              const float* __restrict__ bias, __bf16* __restrict__ C) {
  __shared__ __align__(16) __bf16 at[2][256 * 40];
  __shared__ __align__(16) __bf16 wt[2][64 * 40];
  const int tid    = threadIdx.x;
  const int lane   = tid & 31;
  const int w      = tid >> 5;
  const int m_base = blockIdx.x * 256;
  const int n_base = blockIdx.y * 64;

  // issue this thread's share of one K-chunk copy (A: 4x16B, W: 1x16B)
  auto stage = [&](int buf, int k0) {
#pragma unroll
    for (int i = 0; i < 4; ++i) {
      int c  = tid + (i << 8);
      int m  = c >> 2;
      int kq = (c & 3) * 8;
      async_ld_b128(&at[buf][m * 40 + kq], A,
                    (unsigned)(((m_base + m) * D_MODEL + k0 + kq) * 2));
    }
    int n  = tid >> 2;
    int kq = (tid & 3) * 8;
    async_ld_b128(&wt[buf][n * 40 + kq], W,
                  (unsigned)(((n_base + n) * D_MODEL + k0 + kq) * 2));
  };

  v8f acc[2][4];
  const v8f zacc = {0.f, 0.f, 0.f, 0.f, 0.f, 0.f, 0.f, 0.f};
#pragma unroll
  for (int i = 0; i < 2; ++i)
#pragma unroll
    for (int j = 0; j < 4; ++j) acc[i][j] = zacc;

  stage(0, 0);
  wait_async0();
  __syncthreads();

  int cur = 0;
  for (int k0 = 0; k0 < D_MODEL; k0 += 32) {
    if (k0 + 32 < D_MODEL) stage(cur ^ 1, k0 + 32);   // prefetch next chunk

    v16bf bop[4];
#pragma unroll
    for (int nb = 0; nb < 4; ++nb)
      bop[nb] = load_b32x16(wt[cur], nb * 16, 40, 0, lane);
#pragma unroll
    for (int mi = 0; mi < 2; ++mi) {
      v16bf aop = load_a16x32(at[cur], (w * 2 + mi) * 16, 40, 0, lane);
#pragma unroll
      for (int nb = 0; nb < 4; ++nb)
        acc[mi][nb] = __builtin_amdgcn_wmma_f32_16x16x32_bf16(
            false, aop, false, bop[nb], (short)0, acc[mi][nb], false, false);
    }

    wait_async0();      // next-chunk copies landed
    __syncthreads();    // everyone done reading cur + writes visible
    cur ^= 1;
  }

  // epilogue: + bias, cast bf16, store.  C/D layout: vgpr r -> M = r + 8*(lane>>4)
  const int lr = lane & 15, hi = lane >> 4;
#pragma unroll
  for (int mi = 0; mi < 2; ++mi)
#pragma unroll
    for (int nb = 0; nb < 4; ++nb)
#pragma unroll
      for (int r = 0; r < 8; ++r) {
        int m = m_base + (w * 2 + mi) * 16 + r + 8 * hi;
        int n = n_base + nb * 16 + lr;
        C[(size_t)m * NCOLS + n] = (__bf16)(acc[mi][nb][r] + bias[n]);
      }
}

// ---------------------------------------------------------------------------
// RoPE exactly as the reference applies it: for t of shape (H,B,d,S) it rotates
// pairs (s=2j, s=2j+1), j<16, with angle d * theta^(-j/16).  In-place on bf16 qkv.
__global__ void rope_kernel(__bf16* __restrict__ qkv) {
  int t = blockIdx.x * blockDim.x + threadIdx.x;      // 65536 threads
  int j = t & 15;
  int d = (t >> 4) & 63;
  int h = (t >> 10) & 15;
  int b = t >> 14;
  if (b >= BATCH) return;
  float inv = __expf(-0.57564627324851148f * (float)j);   // ln(10000)/16
  float ang = (float)d * inv;
  float c = cosf(ang), s = sinf(ang);
  size_t base = ((size_t)(b * SEQ + 2 * j)) * NCOLS + h * 192 + d;  // q element
#pragma unroll
  for (int qk = 0; qk < 2; ++qk) {
    size_t p0 = base + qk * 64;            // q then k channel
    float x1 = (float)qkv[p0];
    float x2 = (float)qkv[p0 + NCOLS];     // s+1 row
    qkv[p0]         = (__bf16)(x1 * c - x2 * s);
    qkv[p0 + NCOLS] = (__bf16)(x2 * c + x1 * s);
  }
}

// ---------------------------------------------------------------------------
// Attention core, one workgroup per (h,b):
//   scores[64][64] = q(64xS) . k(64xS)^T * 1/8 ; softmax rows ; out = probs @ v
__global__ __launch_bounds__(256)
void attn_kernel(const __bf16* __restrict__ qkv, float* __restrict__ out) {
  __shared__ __align__(16) __bf16 qt[64 * 36];
  __shared__ __align__(16) __bf16 kt[64 * 36];
  __shared__ __align__(16) float  scs[64 * 65];
  __shared__ __align__(16) __bf16 pr[64 * 66];

  const int tid  = threadIdx.x;
  const int lane = tid & 31;
  const int w    = tid >> 5;
  const int h    = blockIdx.x & 15;
  const int b    = blockIdx.x >> 4;

  const __bf16* qb = qkv + (size_t)b * SEQ * NCOLS + h * 192;  // [s][d], ld=3072
  const __bf16* kb = qb + 64;
  const __bf16* vb = qb + 128;

  // ---- phase 1: scores.  wave w -> d-block w>>1, e half (w&1) (two tiles)
  const int dblk = w >> 1;
  const int e0   = (w & 1) * 32;
  const v8f zacc = {0.f, 0.f, 0.f, 0.f, 0.f, 0.f, 0.f, 0.f};
  v8f s0acc = zacc, s1acc = zacc;

  for (int sc0 = 0; sc0 < SEQ; sc0 += 32) {
    __syncthreads();
    {  // stage q,k 64x32 tiles transposed into [d][s] (s contiguous = K axis)
      int s  = tid >> 3;
      int dq = (tid & 7) * 8;
      uint4 vq = *reinterpret_cast<const uint4*>(qb + (size_t)(sc0 + s) * NCOLS + dq);
      uint4 vk = *reinterpret_cast<const uint4*>(kb + (size_t)(sc0 + s) * NCOLS + dq);
      const __bf16* e  = reinterpret_cast<const __bf16*>(&vq);
      const __bf16* e2 = reinterpret_cast<const __bf16*>(&vk);
#pragma unroll
      for (int i = 0; i < 8; ++i) qt[(dq + i) * 36 + s] = e[i];
#pragma unroll
      for (int i = 0; i < 8; ++i) kt[(dq + i) * 36 + s] = e2[i];
    }
    __syncthreads();
    v16bf aop = load_a16x32(qt, dblk * 16, 36, 0, lane);
    v16bf b0  = load_b32x16(kt, e0,      36, 0, lane);
    v16bf b1  = load_b32x16(kt, e0 + 16, 36, 0, lane);
    s0acc = __builtin_amdgcn_wmma_f32_16x16x32_bf16(false, aop, false, b0,
                                                    (short)0, s0acc, false, false);
    s1acc = __builtin_amdgcn_wmma_f32_16x16x32_bf16(false, aop, false, b1,
                                                    (short)0, s1acc, false, false);
  }

  __syncthreads();
  const int lr = lane & 15, hi = lane >> 4;
#pragma unroll
  for (int r = 0; r < 8; ++r) {
    int d = dblk * 16 + r + 8 * hi;
    scs[d * 65 + e0 + lr]      = s0acc[r] * 0.125f;
    scs[d * 65 + e0 + 16 + lr] = s1acc[r] * 0.125f;
  }
  __syncthreads();

  // ---- softmax over e per row d (threads 0..63)
  if (tid < 64) {
    float mx = -INFINITY;
    for (int e = 0; e < 64; ++e) mx = fmaxf(mx, scs[tid * 65 + e]);
    float sum = 0.f;
    for (int e = 0; e < 64; ++e) {
      float ex = __expf(scs[tid * 65 + e] - mx);
      scs[tid * 65 + e] = ex;
      sum += ex;
    }
    float inv = 1.f / sum;
    for (int e = 0; e < 64; ++e)
      pr[tid * 66 + e] = (__bf16)(scs[tid * 65 + e] * inv);
  }
  __syncthreads();

  // ---- phase 2: out[d][s] = sum_e probs[d][e] v[e][s].  wave w owns 512 s.
  v16bf ap[4][2];
#pragma unroll
  for (int db = 0; db < 4; ++db) {
    ap[db][0] = load_a16x32(pr, db * 16, 66, 0, lane);
    ap[db][1] = load_a16x32(pr, db * 16, 66, 32, lane);
  }
  for (int st = 0; st < 32; ++st) {
    int s0 = w * 512 + st * 16;
    v16bf bv0 = load_bv(vb, s0, 0, lane);
    v16bf bv1 = load_bv(vb, s0, 1, lane);
#pragma unroll
    for (int db = 0; db < 4; ++db) {
      v8f acc = zacc;
      acc = __builtin_amdgcn_wmma_f32_16x16x32_bf16(false, ap[db][0], false, bv0,
                                                    (short)0, acc, false, false);
      acc = __builtin_amdgcn_wmma_f32_16x16x32_bf16(false, ap[db][1], false, bv1,
                                                    (short)0, acc, false, false);
#pragma unroll
      for (int r = 0; r < 8; ++r) {
        int d = db * 16 + r + 8 * hi;
        int s = s0 + lr;
        // reference output permutation: flat = ((h*64+d)*4+b)*4096 + s
        out[((size_t)((h * 64 + d) * 4 + b)) * SEQ + s] = acc[r];
      }
    }
  }
}

// ---------------------------------------------------------------------------
extern "C" void kernel_launch(void* const* d_in, const int* in_sizes, int n_in,
                              void* d_out, int out_size, void* d_ws, size_t ws_size,
                              hipStream_t stream) {
  (void)in_sizes; (void)n_in; (void)out_size; (void)ws_size;
  const float* x    = (const float*)d_in[0];
  const float* W    = (const float*)d_in[1];
  const float* bias = (const float*)d_in[2];
  float* out = (float*)d_out;

  char* ws = (char*)d_ws;
  __bf16* xbf = (__bf16*)ws;                                   // 32 MB
  __bf16* wbf = (__bf16*)(ws + (size_t)MROWS * D_MODEL * 2);   //  6 MB
  __bf16* qkv = (__bf16*)(ws + (size_t)MROWS * D_MODEL * 2
                             + (size_t)NCOLS * D_MODEL * 2);   // 96 MB

  const int nx = MROWS * D_MODEL;
  const int nw = NCOLS * D_MODEL;
  cast_f32_bf16<<<(nx + 255) / 256, 256, 0, stream>>>(x, xbf, nx);
  cast_f32_bf16<<<(nw + 255) / 256, 256, 0, stream>>>(W, wbf, nw);

  dim3 g(MROWS / 256, NCOLS / 64);   // 64 x 48
  qkv_gemm<<<g, 256, 0, stream>>>(xbf, wbf, bias, qkv);

  rope_kernel<<<65536 / 256, 256, 0, stream>>>(qkv);

  attn_kernel<<<N_HEADS * BATCH, 256, 0, stream>>>(qkv, out);
}
